// TurboGL_83580063580809
// MI455X (gfx1250) — 8-bit matrix operand order repaired and checked by reading, compiled, not run on hardware
//
#include <hip/hip_runtime.h>
#include <hip/hip_bf16.h>
#include <stdint.h>

typedef __attribute__((ext_vector_type(16))) int   v16i;
typedef __attribute__((ext_vector_type(8)))  float v8f;

#define T_DIM 4096
#define K_DIM 2048
#define N_DIM 8192
#define E_DIM 8
#define F8MAX 448.0f

#define KSTEP 128
#define NT (K_DIM / KSTEP)
#define A_TILE_BYTES (64 * KSTEP)
#define B_TILE_BYTES (128 * KSTEP)
#define BUF_BYTES (A_TILE_BYTES + B_TILE_BYTES)

__device__ __forceinline__ float bf16_to_f32(unsigned short u) {
  return __uint_as_float(((unsigned)u) << 16);
}

__device__ __forceinline__ int expert_of(int row, const int* lens) {
  int c = 0, e = 0;
#pragma unroll
  for (int i = 0; i < E_DIM; ++i) { c += lens[i]; e += (row >= c) ? 1 : 0; }
  return e;
}

__device__ __forceinline__ unsigned f32_to_e4m3_sw(float f) {
  unsigned s = (__float_as_uint(f) >> 24) & 0x80u;
  float af = fabsf(f);
  if (!(af == af)) return s | 0x7Fu;
  if (af > F8MAX) af = F8MAX;
  if (af < 0.0009765625f) return s;
  unsigned u = __float_as_uint(af);
  int ex = (int)((u >> 23) & 0xFFu) - 127;
  unsigned man = u & 0x7FFFFFu;
  if (ex >= -6) {
    unsigned r = man + 0x7FFFFu + ((man >> 20) & 1u);
    if (r & 0x800000u) { ex++; r = 0; }
    unsigned m3 = (r >> 20) & 7u;
    if (ex > 8) return s | 0x7Eu;
    return s | (unsigned)((ex + 7) << 3) | m3;
  }
  float q = rintf(af * 512.0f);
  unsigned qi = (unsigned)q;
  if (qi >= 8u) return s | 0x08u;
  return s | qi;
}

__device__ __forceinline__ unsigned pack4_e4m3(float x0, float x1, float x2, float x3) {
#if __has_builtin(__builtin_amdgcn_cvt_pk_fp8_f32)
  int v = 0;
  v = __builtin_amdgcn_cvt_pk_fp8_f32(x0, x1, v, false);
  v = __builtin_amdgcn_cvt_pk_fp8_f32(x2, x3, v, true);
  return (unsigned)v;
#else
  return f32_to_e4m3_sw(x0) | (f32_to_e4m3_sw(x1) << 8) |
         (f32_to_e4m3_sw(x2) << 16) | (f32_to_e4m3_sw(x3) << 24);
#endif
}

__device__ __forceinline__ void async_cp16(unsigned lds_off, const unsigned char* g) {
  asm volatile("global_load_async_to_lds_b128 %0, %1, off"
               :: "v"(lds_off), "v"(g) : "memory");
}
__device__ __forceinline__ void wait_async_le6() {
  asm volatile("s_wait_asynccnt 0x6" ::: "memory");
}
__device__ __forceinline__ void wait_async_0() {
  asm volatile("s_wait_asynccnt 0x0" ::: "memory");
}

__global__ void k_amax_a(const unsigned short* __restrict__ a,
                         const int* __restrict__ lens,
                         unsigned* __restrict__ amaxA) {
  __shared__ float red[256];
  const int row = blockIdx.x;
  const unsigned short* p = a + (size_t)row * K_DIM + threadIdx.x * 8;
  float m = 0.f;
#pragma unroll
  for (int i = 0; i < 8; ++i) m = fmaxf(m, fabsf(bf16_to_f32(p[i])));
  red[threadIdx.x] = m;
  __syncthreads();
  for (int s = 128; s > 0; s >>= 1) {
    if (threadIdx.x < (unsigned)s) red[threadIdx.x] = fmaxf(red[threadIdx.x], red[threadIdx.x + s]);
    __syncthreads();
  }
  if (threadIdx.x == 0) atomicMax(&amaxA[expert_of(row, lens)], __float_as_uint(red[0]));
}

__global__ void k_amax_w(const unsigned short* __restrict__ w,
                         unsigned* __restrict__ amaxW) {
  __shared__ float red[256];
  const int row = blockIdx.x;
  const unsigned short* p = w + (size_t)row * K_DIM + threadIdx.x * 8;
  float m = 0.f;
#pragma unroll
  for (int i = 0; i < 8; ++i) m = fmaxf(m, fabsf(bf16_to_f32(p[i])));
  red[threadIdx.x] = m;
  __syncthreads();
  for (int s = 128; s > 0; s >>= 1) {
    if (threadIdx.x < (unsigned)s) red[threadIdx.x] = fmaxf(red[threadIdx.x], red[threadIdx.x + s]);
    __syncthreads();
  }
  if (threadIdx.x == 0) atomicMax(&amaxW[row / N_DIM], __float_as_uint(red[0]));
}

__global__ void k_scales(const unsigned* __restrict__ amaxA,
                         const unsigned* __restrict__ amaxW,
                         float* __restrict__ sA, float* __restrict__ sW) {
  int e = threadIdx.x;
  if (e < E_DIM) {
    sA[e] = fmaxf(__uint_as_float(amaxA[e]), 1e-12f) / F8MAX;
    sW[e] = fmaxf(__uint_as_float(amaxW[e]), 1e-12f) / F8MAX;
  }
}

__global__ void k_quant_a(const unsigned short* __restrict__ a,
                          const int* __restrict__ lens,
                          const float* __restrict__ sA,
                          unsigned char* __restrict__ aq) {
  const size_t idx8 = ((size_t)blockIdx.x * 256 + threadIdx.x) * 8;
  const int row = (int)(idx8 / K_DIM);
  const float inv = 1.0f / sA[expert_of(row, lens)];
  const unsigned short* p = a + idx8;
  float x[8];
#pragma unroll
  for (int i = 0; i < 8; ++i)
    x[i] = fminf(fmaxf(bf16_to_f32(p[i]) * inv, -F8MAX), F8MAX);
  uint2 o;
  o.x = pack4_e4m3(x[0], x[1], x[2], x[3]);
  o.y = pack4_e4m3(x[4], x[5], x[6], x[7]);
  *reinterpret_cast<uint2*>(aq + idx8) = o;
}

__global__ void k_quant_w(const unsigned short* __restrict__ w,
                          const float* __restrict__ sW,
                          unsigned char* __restrict__ wq) {
  const size_t idx8 = ((size_t)blockIdx.x * 256 + threadIdx.x) * 8;
  const int e = (int)(idx8 / ((size_t)N_DIM * K_DIM));
  const float inv = 1.0f / sW[e];
  const unsigned short* p = w + idx8;
  float x[8];
#pragma unroll
  for (int i = 0; i < 8; ++i)
    x[i] = fminf(fmaxf(bf16_to_f32(p[i]) * inv, -F8MAX), F8MAX);
  uint2 o;
  o.x = pack4_e4m3(x[0], x[1], x[2], x[3]);
  o.y = pack4_e4m3(x[4], x[5], x[6], x[7]);
  *reinterpret_cast<uint2*>(wq + idx8) = o;
}

__global__ void __launch_bounds__(256, 2)
k_gemm(const unsigned char* __restrict__ aq,
       const unsigned char* __restrict__ wq,
       const int* __restrict__ lens,
       const float* __restrict__ sA, const float* __restrict__ sW,
       float* __restrict__ out) {
  extern __shared__ unsigned char smem[];

  const int t    = threadIdx.x;
  const int lane = t & 31;
  const int wave = t >> 5;
  const int mBlock = blockIdx.y * 64;
  const int nBlock = blockIdx.x * 128;
  const int mBase  = mBlock + (wave >> 1) * 16;
  const int e = expert_of(mBlock, lens);
  const float scale = sA[e] * sW[e];

  const int rowc = t >> 3;
  const int colb = (t & 7) * 16;
  const unsigned char* gA0 = aq + (size_t)(mBlock + rowc) * K_DIM + colb;
  const unsigned char* gA1 = gA0 + (size_t)32 * K_DIM;
  const unsigned char* gB0 = wq + ((size_t)e * N_DIM + nBlock + rowc) * K_DIM + colb;
  const unsigned lA0 = (unsigned)(rowc * KSTEP + colb);
  const unsigned lA1 = lA0 + 32 * KSTEP;
  const unsigned lB0 = (unsigned)(A_TILE_BYTES + rowc * KSTEP + colb);

  const unsigned aFragOff = (unsigned)(((wave >> 1) * 16 + (lane & 15)) * KSTEP + (lane >> 4) * 16);
  const unsigned bFragCol0 = (unsigned)((wave & 1) * 64 + (lane & 15));
  const unsigned hiB = (unsigned)((lane >> 4) * 16);

  v8f acc[4] = {};

  {
    async_cp16(lA0, gA0);
    async_cp16(lA1, gA1);
#pragma unroll
    for (int i = 0; i < 4; ++i)
      async_cp16(lB0 + (unsigned)(i * 32 * KSTEP), gB0 + (size_t)(i * 32) * K_DIM);
  }

  for (int kt = 0; kt < NT; ++kt) {
    const unsigned base  = (unsigned)((kt & 1) * BUF_BYTES);
    const unsigned nbase = (unsigned)(((kt + 1) & 1) * BUF_BYTES);

    if (kt + 1 < NT) {
      const size_t kOff = (size_t)(kt + 1) * KSTEP;
      async_cp16(nbase + lA0, gA0 + kOff);
      async_cp16(nbase + lA1, gA1 + kOff);
#pragma unroll
      for (int i = 0; i < 4; ++i)
        async_cp16(nbase + lB0 + (unsigned)(i * 32 * KSTEP),
                   gB0 + (size_t)(i * 32) * K_DIM + kOff);
      wait_async_le6();
    } else {
      wait_async_0();
    }
    __syncthreads();

    v16i afr;
    {
      const unsigned char* ap = smem + base + aFragOff;
#pragma unroll
      for (int h = 0; h < 2; ++h)
#pragma unroll
        for (int j = 0; j < 4; ++j) {
          const int2 d = *reinterpret_cast<const int2*>(ap + h * 64 + (j >> 1) * 32 + (j & 1) * 8);
          afr[h * 8 + j * 2]     = d.x;
          afr[h * 8 + j * 2 + 1] = d.y;
        }
    }

    v16i bfr[4];
#pragma unroll
    for (int nt = 0; nt < 4; ++nt) {
      const unsigned char* bp =
          smem + base + A_TILE_BYTES + (bFragCol0 + nt * 16) * KSTEP + hiB;
#pragma unroll
      for (int g = 0; g < 4; ++g) {
        const int4 d = *reinterpret_cast<const int4*>(bp + g * 32);
        bfr[nt][g * 4 + 0] = d.x; bfr[nt][g * 4 + 1] = d.y;
        bfr[nt][g * 4 + 2] = d.z; bfr[nt][g * 4 + 3] = d.w;
      }
    }
#pragma unroll
    for (int nt = 0; nt < 4; ++nt)
      acc[nt] = __builtin_amdgcn_wmma_f32_16x16x128_fp8_fp8(afr, bfr[nt], (short)0,
                                                            acc[nt], false, false);

    __syncthreads();
  }

  const int rowHi = (lane >> 4) * 8;
#pragma unroll
  for (int nt = 0; nt < 4; ++nt) {
    float* op = out + (size_t)(mBase + rowHi) * N_DIM + nBlock + (wave & 1) * 64 +
                nt * 16 + (lane & 15);
#pragma unroll
    for (int r = 0; r < 8; ++r)
      op[(size_t)r * N_DIM] = acc[nt][r] * scale;
  }
}

extern "C" void kernel_launch(void* const* d_in, const int* in_sizes, int n_in,
                              void* d_out, int out_size, void* d_ws, size_t ws_size,
                              hipStream_t stream) {
  (void)in_sizes; (void)n_in; (void)out_size; (void)ws_size;
  const unsigned short* a    = (const unsigned short*)d_in[0];
  const unsigned short* w    = (const unsigned short*)d_in[1];
  const int*            lens = (const int*)d_in[2];
  float* out = (float*)d_out;

  uint8_t* ws = (uint8_t*)d_ws;
  unsigned* amaxA = (unsigned*)(ws + 0);
  unsigned* amaxW = (unsigned*)(ws + 32);
  float*    sA    = (float*)(ws + 64);
  float*    sW    = (float*)(ws + 96);
  unsigned char* aq = ws + 256;
  unsigned char* wq = ws + 256 + (size_t)T_DIM * K_DIM;

  hipMemsetAsync(ws, 0, 128, stream);
  k_amax_a<<<T_DIM, 256, 0, stream>>>(a, lens, amaxA);
  k_amax_w<<<E_DIM * N_DIM, 256, 0, stream>>>(w, amaxW);
  k_scales<<<1, 32, 0, stream>>>(amaxA, amaxW, sA, sW);
  k_quant_a<<<(int)((size_t)T_DIM * K_DIM / 2048), 256, 0, stream>>>(a, lens, sA, aq);
  k_quant_w<<<(int)((size_t)E_DIM * N_DIM * K_DIM / 2048), 256, 0, stream>>>(w, sW, wq);
  k_gemm<<<dim3(N_DIM / 128, T_DIM / 64), 256, 2 * BUF_BYTES, stream>>>(aq, wq, lens,
                                                                        sA, sW, out);
}
